// LastLayersComputation_59828894433321
// MI455X (gfx1250) — matrix operand order repaired and checked by reading, not compiled, not run on hardware
//
#include <hip/hip_runtime.h>

// ---------------------------------------------------------------------------
// ANI2x last-layer MoE ensemble for MI455X (gfx1250, wave32, WMMA).
//
// Pipeline:
//   1) init_hdr      : zero workspace header (counts/cursors)
//   2) shift_kernel  : out[b] = sum_a sae[species[b,a]]   (also inits output)
//   3) count/scan/scatter : bin atoms by species into a compacted index list
//   4) prep_w1       : convert W1 (f32) -> f16 in workspace (halve L2 traffic)
//   5) moe_main      : per 32-atom same-species super-tile x 8 nets:
//                        stage y rows to LDS as f16, v_wmma_f32_16x16x32_f16
//                        over K, +b1, CELU, xW2, lane reduce, atomicAdd out/8
// ---------------------------------------------------------------------------

typedef __attribute__((ext_vector_type(16))) _Float16 v16h;
typedef __attribute__((ext_vector_type(8)))  float    v8f;

union V16H { v16h v; uint4 u[2]; };
union H4   { _Float16 h[4]; uint2 u; };

// ---- workspace header layout (ints) ----
#define HDR_INTS 32
#define H_CNT 0     // [4]  per-species atom counts
#define H_OFF 4     // [5]  per-species offsets into compacted list (+total)
#define H_SB  9     // [5]  per-species super-tile base (+total)
#define H_CUR 14    // [4]  scatter cursors

// ---- LDS staging layout (units: _Float16 elements) ----
// A[g][m][n][k]: k stride KS, per-atom stride MS (8*KS + 8 pad so that the
// per-atom stride is 772 dwords == 4 (mod 64) -> 16 atom rows hit 16 distinct
// 4-bank groups on ds_load_b128), per-tile stride GSZ.
#define KS   192
#define MS   1544          // 8*192 + 8 pad, multiple of 8 (16B alignment)
#define GSZ  (16 * MS)
#define GT   2             // atom tiles (of 16) per block
#define SMEM_BYTES (GT * GSZ * 2 + 256)   // ~97KB f16 staging + atom index tail

// CELU(alpha=0.1): x>0 ? x : 0.1*(exp(10x)-1)
//   = x>0 ? x : fma(exp2(x * 10*log2(e)), 0.1, -0.1)
// Raw v_exp_f32 (no denorm-range fixup) is safe here: for x*SCALE < -126 the
// hw flushes exp2 to 0 and the result is -0.1, the exact CELU asymptote.
#define CELU_EXP2_SCALE 14.426950408889634f   // 10 * log2(e)

// ---------------------------------------------------------------------------
__global__ void init_hdr(int* hdr) {
  if (threadIdx.x < HDR_INTS) hdr[threadIdx.x] = 0;
}

__global__ void shift_kernel(const int* __restrict__ species,
                             const float* __restrict__ sae,
                             float* __restrict__ out, int B) {
  int b = blockIdx.x * blockDim.x + threadIdx.x;
  if (b >= B) return;
  const int* sp = species + (size_t)b * 64;
  float s = 0.f;
  #pragma unroll 8
  for (int a = 0; a < 64; ++a) s += sae[sp[a] & 3];
  out[b] = s;
}

__global__ void count_kernel(const int* __restrict__ species, int* hdr, int BA) {
  int i = blockIdx.x * blockDim.x + threadIdx.x;
  if (i < BA) atomicAdd(&hdr[H_CNT + (species[i] & 3)], 1);
}

__global__ void scan_kernel(int* hdr) {
  if (threadIdx.x == 0 && blockIdx.x == 0) {
    int off = 0, sb = 0;
    for (int s = 0; s < 4; ++s) {
      hdr[H_OFF + s] = off;
      hdr[H_SB + s]  = sb;
      off += hdr[H_CNT + s];
      sb  += (hdr[H_CNT + s] + GT * 16 - 1) / (GT * 16);
    }
    hdr[H_OFF + 4] = off;
    hdr[H_SB + 4]  = sb;
  }
}

__global__ void scatter_kernel(const int* __restrict__ species, int* hdr,
                               int* __restrict__ list, int BA) {
  int i = blockIdx.x * blockDim.x + threadIdx.x;
  if (i < BA) {
    int e = species[i] & 3;
    int pos = hdr[H_OFF + e] + atomicAdd(&hdr[H_CUR + e], 1);
    list[pos] = i;
  }
}

// Convert all W1 weights to f16 (same [n][h][f] layout, concatenated by species)
#define W1SZ0 (8 * 160 * 192)
#define W1SZ2 (8 * 128 * 160)
#define W1OFF1 (W1SZ0)
#define W1OFF2 (2 * W1SZ0)
#define W1OFF3 (2 * W1SZ0 + W1SZ2)
#define W1TOT  (2 * W1SZ0 + 2 * W1SZ2)

__global__ void prep_w1(const float* __restrict__ w0, const float* __restrict__ w1,
                        const float* __restrict__ w2, const float* __restrict__ w3,
                        _Float16* __restrict__ dst) {
  for (int i = blockIdx.x * blockDim.x + threadIdx.x; i < W1TOT;
       i += gridDim.x * blockDim.x) {
    float v;
    if      (i < W1OFF1) v = w0[i];
    else if (i < W1OFF2) v = w1[i - W1OFF1];
    else if (i < W1OFF3) v = w2[i - W1OFF2];
    else                 v = w3[i - W1OFF3];
    dst[i] = (_Float16)v;
  }
}

// ---------------------------------------------------------------------------
// Per-wave compute: 32 atoms (GT tiles of 16) x one net x (H/2 hidden columns), WMMA over K.
// A fragment (f16 16x32): lane L -> row L&15; element j -> K = j + 16*(L>=16)
// B fragment (f16 32x16): lane L -> col L&15; element j -> K = j + 16*(L>=16)
// D/C (f32 16x16):        lane L -> col L&15; reg r -> row r + 8*(L>=16)
// ---------------------------------------------------------------------------
template <int F, int H>
__device__ __forceinline__ void tile_compute(
    const _Float16* __restrict__ Alds, const int* __restrict__ sAtom,
    const _Float16* __restrict__ w1h,   // W1h base for (species, net n): [H][F] f16
    const float* __restrict__ b1p, const float* __restrict__ w2p, float b2v,
    float* __restrict__ out, int n, int halfSel, int lane) {
  constexpr int KSTEPS = F / 32;
  constexpr int HALFT  = (H / 16) / 2;
  const int laneN = lane & 15;
  const int hi    = lane >> 4;

  float contrib[GT][8];
  #pragma unroll
  for (int g = 0; g < GT; ++g)
    #pragma unroll
    for (int r = 0; r < 8; ++r) contrib[g][r] = 0.f;

  #pragma unroll
  for (int it = 0; it < HALFT; ++it) {
    const int t   = halfSel * HALFT + it;
    const int col = t * 16 + laneN;

    // Load B fragments for all k-steps (held in VGPRs across both atom tiles)
    V16H Bf[KSTEPS];
    const _Float16* bp = w1h + (size_t)col * F + 16 * hi;
    #pragma unroll
    for (int s6 = 0; s6 < KSTEPS; ++s6) {
      Bf[s6].u[0] = *reinterpret_cast<const uint4*>(bp + 32 * s6);
      Bf[s6].u[1] = *reinterpret_cast<const uint4*>(bp + 32 * s6 + 8);
    }

    const float b1v = b1p[n * H + col];
    const float w2v = w2p[n * H + col];

    #pragma unroll
    for (int g = 0; g < GT; ++g) {
      v8f acc = {0.f, 0.f, 0.f, 0.f, 0.f, 0.f, 0.f, 0.f};
      const int abase = g * GSZ + laneN * MS + n * KS + 16 * hi;
      #pragma unroll
      for (int s6 = 0; s6 < KSTEPS; ++s6) {
        V16H Af;
        Af.u[0] = *reinterpret_cast<const uint4*>(&Alds[abase + 32 * s6]);
        Af.u[1] = *reinterpret_cast<const uint4*>(&Alds[abase + 32 * s6 + 8]);
        acc = __builtin_amdgcn_wmma_f32_16x16x32_f16(
            /*neg_a=*/false, Af.v, /*neg_b=*/false, Bf[s6].v,
            /*c_mod=*/(short)0, acc, /*reuse_a=*/false, /*reuse_b=*/false);
      }
      // epilogue: +b1, CELU(alpha=0.1), x W2, accumulate over column tiles.
      // Raw v_exp_f32 path: mul, exp, fma, cmp, cndmask, fma = 6 VALU/elt,
      // co-executes with the WMMA pipe (XDL ops tracked as TRANS).
      #pragma unroll
      for (int r = 0; r < 8; ++r) {
        const float hx  = acc[r] + b1v;
        const float ex  = __builtin_amdgcn_exp2f(hx * CELU_EXP2_SCALE);
        const float neg = __builtin_fmaf(ex, 0.1f, -0.1f);
        const float act = (hx > 0.f) ? hx : neg;
        contrib[g][r] = __builtin_fmaf(act, w2v, contrib[g][r]);
      }
    }
  }

  // Reduce over the 16 hidden columns (lanes 0-15 hold rows 0-7, 16-31 rows 8-15)
  #pragma unroll
  for (int g = 0; g < GT; ++g)
    #pragma unroll
    for (int r = 0; r < 8; ++r) {
      float red = contrib[g][r];
      red += __shfl_xor(red, 1);
      red += __shfl_xor(red, 2);
      red += __shfl_xor(red, 4);
      red += __shfl_xor(red, 8);
      contrib[g][r] = red;
    }

  if (laneN == 0) {
    const float bb = (halfSel == 0) ? b2v : 0.f;   // add b2 once per (atom,net)
    #pragma unroll
    for (int g = 0; g < GT; ++g)
      #pragma unroll
      for (int r = 0; r < 8; ++r) {
        const int m  = r + 8 * hi;
        const int af = sAtom[g * 16 + m];
        if (af >= 0)
          atomicAdd(&out[af >> 6], (contrib[g][r] + bb) * 0.125f); // mean over 8 nets
      }
  }
}

// ---------------------------------------------------------------------------
__global__ __launch_bounds__(512, 1)
void moe_main(const float* __restrict__ y,
              const float* __restrict__ b1_0, const float* __restrict__ b1_1,
              const float* __restrict__ b1_2, const float* __restrict__ b1_3,
              const float* __restrict__ W2_0, const float* __restrict__ W2_1,
              const float* __restrict__ W2_2, const float* __restrict__ W2_3,
              const float* __restrict__ b2_0, const float* __restrict__ b2_1,
              const float* __restrict__ b2_2, const float* __restrict__ b2_3,
              const int* __restrict__ hdr, const int* __restrict__ list,
              const _Float16* __restrict__ W1h, float* __restrict__ out) {
  extern __shared__ char smem[];
  _Float16* Alds = reinterpret_cast<_Float16*>(smem);
  int* sAtom = reinterpret_cast<int*>(smem + GT * GSZ * 2);

  const int blk = blockIdx.x;
  if (blk >= hdr[H_SB + 4]) return;
  int s = 0;
  while (s < 3 && blk >= hdr[H_SB + s + 1]) ++s;
  const int superIdx = blk - hdr[H_SB + s];
  const int cnt  = hdr[H_CNT + s];
  const int base = hdr[H_OFF + s];
  const int tid  = threadIdx.x;

  if (tid < GT * 16) {
    const int p = superIdx * (GT * 16) + tid;
    sAtom[tid] = (p < cnt) ? list[base + p] : -1;
  }
  __syncthreads();

  const int F = (s < 2) ? 192 : 160;
  const int safeAtom = sAtom[0];   // always valid (block exists => >=1 atom)

  // Stage y rows -> LDS as f16 (coalesced float4 reads, 97KB per block)
  const int chunksPerRow = F / 4;
  const int totalChunks  = GT * 16 * 8 * chunksPerRow;
  for (int c = tid; c < totalChunks; c += blockDim.x) {
    const int row = c / chunksPerRow;
    const int kc  = c - row * chunksPerRow;
    const int nn  = row & 7;
    const int mm  = (row >> 3) & 15;
    const int gg  = row >> 7;
    int af = sAtom[gg * 16 + mm];
    if (af < 0) af = safeAtom;     // pad rows: load valid data, mask at the add
    const float4 f = *reinterpret_cast<const float4*>(
        y + (size_t)af * (8 * 192) + nn * 192 + kc * 4);
    H4 h4;
    h4.h[0] = (_Float16)f.x; h4.h[1] = (_Float16)f.y;
    h4.h[2] = (_Float16)f.z; h4.h[3] = (_Float16)f.w;
    *reinterpret_cast<uint2*>(&Alds[gg * GSZ + mm * MS + nn * KS + kc * 4]) = h4.u;
  }
  __syncthreads();

  const int lane    = tid & 31;
  const int w       = tid >> 5;   // 16 waves
  const int n       = w & 7;      // ensemble net
  const int halfSel = w >> 3;     // column half

  if (s == 0) {
    tile_compute<192, 160>(Alds, sAtom, W1h + (size_t)n * 160 * 192,
                           b1_0, W2_0, b2_0[n], out, n, halfSel, lane);
  } else if (s == 1) {
    tile_compute<192, 160>(Alds, sAtom, W1h + W1OFF1 + (size_t)n * 160 * 192,
                           b1_1, W2_1, b2_1[n], out, n, halfSel, lane);
  } else if (s == 2) {
    tile_compute<160, 128>(Alds, sAtom, W1h + W1OFF2 + (size_t)n * 128 * 160,
                           b1_2, W2_2, b2_2[n], out, n, halfSel, lane);
  } else {
    tile_compute<160, 128>(Alds, sAtom, W1h + W1OFF3 + (size_t)n * 128 * 160,
                           b1_3, W2_3, b2_3[n], out, n, halfSel, lane);
  }
}

// ---------------------------------------------------------------------------
extern "C" void kernel_launch(void* const* d_in, const int* in_sizes, int n_in,
                              void* d_out, int out_size, void* d_ws, size_t ws_size,
                              hipStream_t stream) {
  const int*   species = (const int*)d_in[0];
  const float* y       = (const float*)d_in[1];
  const float* W1p[4]  = {(const float*)d_in[2],  (const float*)d_in[6],
                          (const float*)d_in[10], (const float*)d_in[14]};
  const float* b1p[4]  = {(const float*)d_in[3],  (const float*)d_in[7],
                          (const float*)d_in[11], (const float*)d_in[15]};
  const float* W2p[4]  = {(const float*)d_in[4],  (const float*)d_in[8],
                          (const float*)d_in[12], (const float*)d_in[16]};
  const float* b2p[4]  = {(const float*)d_in[5],  (const float*)d_in[9],
                          (const float*)d_in[13], (const float*)d_in[17]};
  const float* sae     = (const float*)d_in[18];
  float* out = (float*)d_out;

  const int BA = in_sizes[0];      // B*A = 262144
  const int B  = out_size;         // 4096

  // workspace: [header ints][compacted atom list][W1 f16]
  int* hdr  = (int*)d_ws;
  int* list = hdr + HDR_INTS;
  _Float16* W1h = (_Float16*)((char*)d_ws + (size_t)(HDR_INTS + BA) * sizeof(int));

  init_hdr<<<1, 64, 0, stream>>>(hdr);
  shift_kernel<<<(B + 127) / 128, 128, 0, stream>>>(species, sae, out, B);
  count_kernel<<<(BA + 255) / 256, 256, 0, stream>>>(species, hdr, BA);
  scan_kernel<<<1, 32, 0, stream>>>(hdr);
  scatter_kernel<<<(BA + 255) / 256, 256, 0, stream>>>(species, hdr, list, BA);
  prep_w1<<<512, 256, 0, stream>>>(W1p[0], W1p[1], W1p[2], W1p[3], W1h);

  // ~97KB dynamic LDS (gfx1250 WGP has 320KB); opt in above the default cap.
  (void)hipFuncSetAttribute(reinterpret_cast<const void*>(moe_main),
                            hipFuncAttributeMaxDynamicSharedMemorySize, SMEM_BYTES);
  const int maxBlocks = BA / (GT * 16) + 4;   // upper bound on super-tiles
  moe_main<<<maxBlocks, 512, SMEM_BYTES, stream>>>(
      y, b1p[0], b1p[1], b1p[2], b1p[3],
      W2p[0], W2p[1], W2p[2], W2p[3],
      b2p[0], b2p[1], b2p[2], b2p[3],
      hdr, list, W1h, out);
}
